// LorentzMLP_87668872446024
// MI455X (gfx1250) — matrix operand order repaired and checked by reading, not compiled, not run on hardware
//
#include <hip/hip_runtime.h>
#include <hip/hip_bf16.h>
#include <stdint.h>

// Lorentz-MLP for MI455X (gfx1250, wave32).
// Split-fp32-into-bf16 GEMMs on v_wmma_f32_16x16x32_bf16 (hi*hi + hi*lo + lo*hi,
// f32 accumulate -> ~fp32 accuracy). Conversions hoisted into bandwidth-bound
// elementwise kernels. GEMM tiles are streamed with CDNA5 async loads directly
// into LDS (global_load_async_to_lds_b128 + s_wait_asynccnt) with 2-deep LDS
// double buffering, so the inner loop is: wait/barrier -> issue next async tile
// -> ds_load_b128 fragments -> 24 WMMAs. No VGPR staging, no ds_store.
// Workspace: Y f32 (134MB) + tangent hi/lo planes (134MB) + weight hi/lo planes
// (67MB, reused per layer) = 336MB.

#define EPSF 1e-7f
#define MAXTAN 10.0f

typedef __attribute__((ext_vector_type(16))) __bf16 v16bf;
typedef __attribute__((ext_vector_type(8)))  float  v8f;
typedef __attribute__((ext_vector_type(4)))  float  f32x4;
typedef __attribute__((ext_vector_type(4)))  unsigned u32x4;
typedef __attribute__((ext_vector_type(2)))  unsigned u32x2;

#define TM 128     // block tile rows (activation rows)
#define TN 128     // block tile cols (output features)
#define BK 32      // K per WMMA step / LDS stage
#define AST 40     // LDS row stride in bf16 elements (32 + 8 pad; 80B, 16B-aligned)
#define PLE (TM * AST)        // plane size in elements
#define PLB (PLE * 2)         // plane size in bytes (10240)
#define BUFB (4 * PLB)        // one double-buffer slot in bytes (40960)

__device__ __forceinline__ unsigned bf16_rne(float f) {
    unsigned u = __builtin_bit_cast(unsigned, f);
    return (u + 0x7FFFu + ((u >> 16) & 1u)) >> 16;
}
__device__ __forceinline__ float bf16_to_f(unsigned h) {
    return __builtin_bit_cast(float, h << 16);
}
__device__ __forceinline__ void split_bf16(float v, unsigned& h, unsigned& l) {
    h = bf16_rne(v);
    l = bf16_rne(v - bf16_to_f(h));
}

__device__ __forceinline__ v8f wmma_bf16(v16bf a, v16bf b, v8f c) {
    return __builtin_amdgcn_wmma_f32_16x16x32_bf16(
        /*neg_a=*/false, a, /*neg_b=*/false, b,
        /*c_mod=*/(short)0, c, /*reuse_a=*/false, /*reuse_b=*/false);
}

// CDNA5 async copy: 16B per lane, global (SGPR base + 32-bit VGPR offset, GVS)
// -> LDS (32-bit LDS byte address in VGPR). Tracked by ASYNCcnt.
__device__ __forceinline__ void async_ld_b128(unsigned lds, unsigned voff, const void* base) {
    asm volatile("global_load_async_to_lds_b128 %0, %1, %2 offset:0"
                 :: "v"(lds), "v"(voff), "s"(base) : "memory");
}
__device__ __forceinline__ void wait_async0() {
    asm volatile("s_wait_asynccnt 0x0" ::: "memory");
}

// A fragment (16x32 bf16, MxK): lane r=lane&15 is row M=r; k0=(lane>>4)*8.
// VGPR0-3 hold K=k0..k0+7 (packed pairs), VGPR4-7 hold K=16+k0..16+k0+7.
__device__ __forceinline__ v16bf load_frag_a(const unsigned short* base, int lane) {
    int r = lane & 15;
    int k0 = (lane >> 4) * 8;
    const unsigned short* p = base + r * AST;
    union U { u32x4 q[2]; v16bf v; } u;
    u.q[0] = *(const u32x4*)(p + k0);
    u.q[1] = *(const u32x4*)(p + 16 + k0);
    return u.v;
}

// B fragment (32x16 bf16, KxN): lane n=lane&15 is column N=n; kb=(lane>>4)*8.
// B[k][n] = W[colBase+n][k]: each lane reads K=kb..kb+7 and K=16+kb..16+kb+7 of one W row.
__device__ __forceinline__ v16bf load_frag_b(const unsigned short* base, int lane) {
    int r = lane & 15;
    int kb = (lane >> 4) * 8;
    const unsigned short* p = base + r * AST + kb;
    union U { u32x4 q[2]; v16bf v; } u;
    u.q[0] = *(const u32x4*)(p);
    u.q[1] = *(const u32x4*)(p + 16);
    return u.v;
}

// Y[i, m] = sum_k A[i,k] * W[m,k] + bias[m], all operands pre-split bf16 planes.
// Ahi/Alo: NxK (ld=K), Whi/Wlo: MxK (ld=K), Y: NxM f32. K%32==0, tiles exact.
__global__ __launch_bounds__(256) void lorentz_gemm_bf16x3(
    const unsigned short* __restrict__ Ahi, const unsigned short* __restrict__ Alo,
    const unsigned short* __restrict__ Whi, const unsigned short* __restrict__ Wlo,
    const float* __restrict__ bias, float* __restrict__ Y,
    int K, int M)
{
    // 2 buffers x 4 planes (Ahi,Alo,Bhi,Blo) x 128x40 bf16 = 80 KB
    __shared__ unsigned short smem[8 * PLE];

    const int tid  = threadIdx.x;
    const int lane = tid & 31;
    const int w    = tid >> 5;
    const int wm   = (w & 3) * 32;   // 4 wave-rows of 32
    const int wn   = (w >> 2) * 64;  // 2 wave-cols of 64
    const int rowBase = blockIdx.y * TM;
    const int colBase = blockIdx.x * TN;

    const v8f vzero = {0.f, 0.f, 0.f, 0.f, 0.f, 0.f, 0.f, 0.f};
    v8f acc[2][4];
#pragma unroll
    for (int i = 0; i < 2; ++i)
#pragma unroll
        for (int j = 0; j < 4; ++j) acc[i][j] = vzero;

    // per-thread staging coordinates: 512 16B-chunks per plane, 2 per thread
    // chunk idx = tid + p*256 -> row = idx>>2 (0..127), c8 = idx&3 (8-elem chunk)
    int r0 = tid >> 2,           c80 = tid & 3;
    int r1 = (tid + 256) >> 2,   c81 = (tid + 256) & 3;
    // LDS byte addresses (flat-pointer low 32 bits == LDS byte address)
    unsigned lds0 = (unsigned)(uintptr_t)&smem[r0 * AST + c80 * 8];
    unsigned lds1 = (unsigned)(uintptr_t)&smem[r1 * AST + c81 * 8];
    // global byte offsets from plane bases (fit in 32 bits: buffers <= 134MB)
    unsigned voffA0 = (unsigned)(((size_t)(rowBase + r0) * K + c80 * 8) * 2);
    unsigned voffA1 = (unsigned)(((size_t)(rowBase + r1) * K + c81 * 8) * 2);
    unsigned voffB0 = (unsigned)(((size_t)(colBase + r0) * K + c80 * 8) * 2);
    unsigned voffB1 = (unsigned)(((size_t)(colBase + r1) * K + c81 * 8) * 2);

    const int NT = K / BK;

    // prologue: fill buffer 0 asynchronously
    {
        async_ld_b128(lds0,           voffA0, Ahi);
        async_ld_b128(lds0 + PLB,     voffA0, Alo);
        async_ld_b128(lds0 + 2 * PLB, voffB0, Whi);
        async_ld_b128(lds0 + 3 * PLB, voffB0, Wlo);
        async_ld_b128(lds1,           voffA1, Ahi);
        async_ld_b128(lds1 + PLB,     voffA1, Alo);
        async_ld_b128(lds1 + 2 * PLB, voffB1, Whi);
        async_ld_b128(lds1 + 3 * PLB, voffB1, Wlo);
        voffA0 += BK * 2; voffA1 += BK * 2;
        voffB0 += BK * 2; voffB1 += BK * 2;
    }

    for (int kt = 0; kt < NT; ++kt) {
        wait_async0();     // this wave's current-buffer async loads have landed
        __syncthreads();   // all waves: cur buffer filled; other buffer free

        if (kt + 1 < NT) {  // stream next tile into the other buffer (overlaps WMMA)
            unsigned bo = (unsigned)(((kt + 1) & 1) * BUFB);
            async_ld_b128(lds0 + bo,           voffA0, Ahi);
            async_ld_b128(lds0 + bo + PLB,     voffA0, Alo);
            async_ld_b128(lds0 + bo + 2 * PLB, voffB0, Whi);
            async_ld_b128(lds0 + bo + 3 * PLB, voffB0, Wlo);
            async_ld_b128(lds1 + bo,           voffA1, Ahi);
            async_ld_b128(lds1 + bo + PLB,     voffA1, Alo);
            async_ld_b128(lds1 + bo + 2 * PLB, voffB1, Whi);
            async_ld_b128(lds1 + bo + 3 * PLB, voffB1, Wlo);
            voffA0 += BK * 2; voffA1 += BK * 2;
            voffB0 += BK * 2; voffB1 += BK * 2;
        }

        const unsigned short* sb    = smem + (kt & 1) * (4 * PLE);
        const unsigned short* As_hi = sb;
        const unsigned short* As_lo = sb + PLE;
        const unsigned short* Bs_hi = sb + 2 * PLE;
        const unsigned short* Bs_lo = sb + 3 * PLE;

        v16bf ah[2], al[2], bh[4], bl[4];
#pragma unroll
        for (int i = 0; i < 2; ++i) {
            ah[i] = load_frag_a(As_hi + (wm + 16 * i) * AST, lane);
            al[i] = load_frag_a(As_lo + (wm + 16 * i) * AST, lane);
        }
#pragma unroll
        for (int j = 0; j < 4; ++j) {
            bh[j] = load_frag_b(Bs_hi + (wn + 16 * j) * AST, lane);
            bl[j] = load_frag_b(Bs_lo + (wn + 16 * j) * AST, lane);
        }
#pragma unroll
        for (int i = 0; i < 2; ++i)
#pragma unroll
            for (int j = 0; j < 4; ++j) {
                v8f c = acc[i][j];
                c = wmma_bf16(ah[i], bh[j], c);  // hi*hi
                c = wmma_bf16(ah[i], bl[j], c);  // hi*lo
                c = wmma_bf16(al[i], bh[j], c);  // lo*hi
                acc[i][j] = c;
            }
    }

    // store: 32-bit C/D 16x16 layout: VGPR j -> M=j (lanes 0-15) / M=8+j (16-31), N=lane&15
    const int lr = (lane >> 4) * 8;
    const int lc = lane & 15;
#pragma unroll
    for (int j = 0; j < 4; ++j) {
        int col = colBase + wn + 16 * j + lc;
        float bsv = bias[col];
#pragma unroll
        for (int i = 0; i < 2; ++i) {
            int row0 = rowBase + wm + 16 * i + lr;
            float* yp = Y + (size_t)row0 * M + col;
#pragma unroll
            for (int jj = 0; jj < 8; ++jj)
                yp[(size_t)jj * M] = acc[i][j][jj] + bsv;
        }
    }
}

// ---------- f32 -> (bf16 hi, bf16 lo) plane conversion (weights) ----------
__global__ __launch_bounds__(256) void cvt_f32_to_bf16x2(
    const float* __restrict__ src,
    unsigned short* __restrict__ hi, unsigned short* __restrict__ lo, int n4)
{
    int stride = gridDim.x * blockDim.x;
    for (int i = blockIdx.x * blockDim.x + threadIdx.x; i < n4; i += stride) {
        f32x4 v = *(const f32x4*)(src + (size_t)i * 4);
        unsigned h0, h1, h2, h3, l0, l1, l2, l3;
        split_bf16(v.x, h0, l0);
        split_bf16(v.y, h1, l1);
        split_bf16(v.z, h2, l2);
        split_bf16(v.w, h3, l3);
        u32x2 ph = {h0 | (h1 << 16), h2 | (h3 << 16)};
        u32x2 pl = {l0 | (l1 << 16), l2 | (l3 << 16)};
        *(u32x2*)(hi + (size_t)i * 4) = ph;
        *(u32x2*)(lo + (size_t)i * 4) = pl;
    }
}

// ---------- row-wise hyperbolic epilogues (one block per row) ----------

__device__ __forceinline__ float block_reduce_ssq(float v, float* red) {
    red[threadIdx.x] = v;
    __syncthreads();
    for (int s = 128; s > 0; s >>= 1) {
        if (threadIdx.x < (unsigned)s) red[threadIdx.x] += red[threadIdx.x + s];
        __syncthreads();
    }
    return red[0];
}

__device__ __forceinline__ void store_split(unsigned short* hi, unsigned short* lo, float v) {
    unsigned h, l;
    split_bf16(v, h, l);
    *hi = (unsigned short)h;
    *lo = (unsigned short)l;
}

// t = logmap0(x_hyp), written as bf16 hi/lo planes (ld = D)
__global__ __launch_bounds__(256) void lorentz_tangent_in(
    const float* __restrict__ x,
    unsigned short* __restrict__ thi, unsigned short* __restrict__ tlo, int D)
{
    __shared__ float red[256];
    const int row = blockIdx.x;
    const float* xr = x + (size_t)row * D;
    unsigned short* th = thi + (size_t)row * D;
    unsigned short* tl = tlo + (size_t)row * D;
    float ssq = 0.f;
    for (int i = 1 + threadIdx.x; i < D; i += 256) { float v = xr[i]; ssq += v * v; }
    ssq = block_reduce_ssq(ssq, red);
    float ns = fmaxf(sqrtf(ssq), EPSF);
    float d  = acoshf(fmaxf(xr[0], 1.0f + EPSF));
    float s  = d / ns;
    if (threadIdx.x == 0) { th[0] = 0; tl[0] = 0; }
    for (int i = 1 + threadIdx.x; i < D; i += 256) store_split(th + i, tl + i, xr[i] * s);
}

// t_next = logmap0(projx(expmap0(logmap0(projx(expmap0(y)))))), per-row scalar chain
__global__ __launch_bounds__(256) void lorentz_mid(
    const float* __restrict__ y,
    unsigned short* __restrict__ thi, unsigned short* __restrict__ tlo, int D)
{
    __shared__ float red[256];
    const int row = blockIdx.x;
    const float* yr = y + (size_t)row * D;
    unsigned short* th = thi + (size_t)row * D;
    unsigned short* tl = tlo + (size_t)row * D;
    float ssq = 0.f;
    for (int i = 1 + threadIdx.x; i < D; i += 256) { float v = yr[i]; ssq += v * v; }
    ssq = block_reduce_ssq(ssq, red);

    // expmap0 + projx
    float n1  = fmaxf(sqrtf(ssq), EPSF);
    float nc1 = fminf(n1, MAXTAN);
    float s1  = sinhf(nc1) / n1;            // y[1:] -> xs
    float r1  = s1 * n1;                    // ||xs||
    float x0  = sqrtf(1.0f + r1 * r1);
    // logmap0
    float d1  = acoshf(fmaxf(x0, 1.0f + EPSF));
    float s2  = d1 / fmaxf(r1, EPSF);
    // expmap0 + projx (second round trip)
    float n2  = fmaxf(s1 * s2 * n1, EPSF);
    float nc2 = fminf(n2, MAXTAN);
    float s3  = sinhf(nc2) / n2;
    float r2  = s3 * n2;
    float x0b = sqrtf(1.0f + r2 * r2);
    // logmap0 (feeds next layer's tangent)
    float d2  = acoshf(fmaxf(x0b, 1.0f + EPSF));
    float s4  = d2 / fmaxf(r2, EPSF);

    float S = s1 * s2 * s3 * s4;
    if (threadIdx.x == 0) { th[0] = 0; tl[0] = 0; }
    for (int i = 1 + threadIdx.x; i < D; i += 256) store_split(th + i, tl + i, yr[i] * S);
}

// out = projx(safe_expmap0(y)) in f32
__global__ __launch_bounds__(256) void lorentz_out(
    const float* __restrict__ y, float* __restrict__ o, int D)
{
    __shared__ float red[256];
    const int row = blockIdx.x;
    const float* yr = y + (size_t)row * D;
    float* orow = o + (size_t)row * D;
    float ssq = 0.f;
    for (int i = 1 + threadIdx.x; i < D; i += 256) { float v = yr[i]; ssq += v * v; }
    ssq = block_reduce_ssq(ssq, red);
    float n  = fmaxf(sqrtf(ssq), EPSF);
    float nc = fminf(n, MAXTAN);
    float s  = sinhf(nc) / n;
    float x0 = sqrtf(1.0f + s * s * ssq);
    if (threadIdx.x == 0) orow[0] = x0;
    for (int i = 1 + threadIdx.x; i < D; i += 256) orow[i] = yr[i] * s;
}

extern "C" void kernel_launch(void* const* d_in, const int* in_sizes, int n_in,
                              void* d_out, int out_size, void* d_ws, size_t ws_size,
                              hipStream_t stream) {
    const int N = 8192, D_IN = 1024, D_HID = 4096, D_OUT = 1024;
    const float* x  = (const float*)d_in[0];
    const float* W1 = (const float*)d_in[1];
    const float* b1 = (const float*)d_in[2];
    const float* W2 = (const float*)d_in[3];
    const float* b2 = (const float*)d_in[4];
    const float* W3 = (const float*)d_in[5];
    const float* b3 = (const float*)d_in[6];
    float* out = (float*)d_out;

    // workspace layout (336 MB total)
    char* p = (char*)d_ws;
    float* Y = (float*)p;                       p += (size_t)N * D_HID * 4;   // 134MB
    unsigned short* Thi = (unsigned short*)p;   p += (size_t)N * D_HID * 2;   // 67MB
    unsigned short* Tlo = (unsigned short*)p;   p += (size_t)N * D_HID * 2;   // 67MB
    unsigned short* Whi = (unsigned short*)p;   p += (size_t)D_HID * D_HID * 2; // 33.5MB
    unsigned short* Wlo = (unsigned short*)p;                                  // 33.5MB

    dim3 blk(256);
    dim3 cvg(4096);

    // layer 1
    lorentz_tangent_in<<<N, blk, 0, stream>>>(x, Thi, Tlo, D_IN);
    cvt_f32_to_bf16x2<<<cvg, blk, 0, stream>>>(W1, Whi, Wlo, D_HID * D_IN / 4);
    lorentz_gemm_bf16x3<<<dim3(D_HID / TN, N / TM), blk, 0, stream>>>(
        Thi, Tlo, Whi, Wlo, b1, Y, D_IN, D_HID);
    lorentz_mid<<<N, blk, 0, stream>>>(Y, Thi, Tlo, D_HID);
    // layer 2
    cvt_f32_to_bf16x2<<<cvg, blk, 0, stream>>>(W2, Whi, Wlo, D_HID * D_HID / 4);
    lorentz_gemm_bf16x3<<<dim3(D_HID / TN, N / TM), blk, 0, stream>>>(
        Thi, Tlo, Whi, Wlo, b2, Y, D_HID, D_HID);
    lorentz_mid<<<N, blk, 0, stream>>>(Y, Thi, Tlo, D_HID);
    // layer 3
    cvt_f32_to_bf16x2<<<cvg, blk, 0, stream>>>(W3, Whi, Wlo, D_OUT * D_HID / 4);
    lorentz_gemm_bf16x3<<<dim3(D_OUT / TN, N / TM), blk, 0, stream>>>(
        Thi, Tlo, Whi, Wlo, b3, Y, D_HID, D_OUT);
    lorentz_out<<<N, blk, 0, stream>>>(Y, out, D_OUT);
}
